// TACE_33414845563089
// MI455X (gfx1250) — matrix operand order repaired and checked by reading, not compiled, not run on hardware
//
#include <hip/hip_runtime.h>
#include <hip/hip_bf16.h>

typedef __attribute__((ext_vector_type(16))) _Float16 v16h;
typedef __attribute__((ext_vector_type(8)))  _Float16 v8h;
typedef __attribute__((ext_vector_type(8)))  float    v8f;

#define WMMA_F16(a,b,c) __builtin_amdgcn_wmma_f32_16x16x32_f16(false,(a),false,(b),(short)0,(c),false,false)

#define C_CH 64
#define DIMS 40   // 1+3+9+27

// swizzled per-layer weight bank layout (halfs):
//   [w1s 2048][w2s 4096][w3s 4096][w4s 16384][mixs 16384]  = 43008 per layer
#define W1_OFF   0
#define W2_OFF   2048
#define W3_OFF   6144
#define W4_OFF   10240
#define MIX_OFF  26624
#define PER_L    43008

__device__ __forceinline__ float silu_f(float x){
  return x * __builtin_amdgcn_rcpf(1.f + __expf(-x));   // 1 trans + 1 rcp, no IEEE div
}

__device__ __forceinline__ v16h join8(v8h lo, v8h hi){
  return __builtin_shufflevector(lo, hi, 0,1,2,3,4,5,6,7,8,9,10,11,12,13,14,15);
}

// ---- WMMA fragment loaders (wave32, 16x16x32 f16) ----
// A 16x32 f16, row-major tile in LDS, ld in halfs (16B-aligned rows).
// lane%16 = row M; lanes 0-15 hold K 0-7 & 16-23, lanes 16-31 hold K 8-15 & 24-31.
__device__ __forceinline__ v16h fragA_lds(const _Float16* p, int ld, int k0, int lane){
  const _Float16* r = p + (lane & 15)*ld + k0 + ((lane >> 4) << 3);
  v8h lo = *(const v8h*)r;          // ds_load_b128
  v8h hi = *(const v8h*)(r + 16);   // ds_load_b128
  return join8(lo, hi);
}
// A 16x32 from global f16 [rows][256] (row pointer pre-resolved, 16B aligned).
__device__ __forceinline__ v16h fragA_g16(const _Float16* p, int k0, int lane){
  const _Float16* r = p + k0 + ((lane >> 4) << 3);
  v8h lo = *(const v8h*)r;          // global_load_b128
  v8h hi = *(const v8h*)(r + 16);
  return join8(lo, hi);
}
// B 32x16: fragment-major pre-swizzled bank; each lane's 16 halfs contiguous (32B).
__device__ __forceinline__ v16h fragB_sw(const _Float16* base, int frag, int lane){
  const v8h* p = (const v8h*)(base + (size_t)((frag << 5) + lane)*16);
  return join8(p[0], p[1]);         // 2x global_load_b128 (L2-resident)
}

// ============== prep: swizzle all B weights to fragment-major f16 ==============
// dst element for (k, col) of a [K][NCOLS] row-major matrix:
//   nt=col/16, n=col%16, kstep=k/32, kg=(k/8)%2, s=k%8 + 8*((k/16)%2)
//   idx = ((kstep*NT + nt)*32 + kg*16 + n)*16 + s
__global__ void tace_prep(const float* __restrict__ rw1, const float* __restrict__ rw2,
                          const float* __restrict__ rw3, const float* __restrict__ rw4,
                          const float* __restrict__ mixw, _Float16* __restrict__ bank){
  int t = blockIdx.x*blockDim.x + threadIdx.x;
  if (t >= 2*PER_L) return;
  int l = t / PER_L, r = t - l*PER_L;
  _Float16* base = bank + (size_t)l*PER_L;
  int k, c, NT; float val; _Float16* out;
  if (r < 2048){                       // w1: [32 (pad)][64]
    k = r >> 6; c = r & 63; NT = 4;
    val = (k < 8) ? rw1[l*512 + k*64 + c] : 0.f;
    out = base + W1_OFF;
  } else if (r < 6144){                // w2: [64][64]
    int q = r - 2048; k = q >> 6; c = q & 63; NT = 4;
    val = rw2[l*4096 + q]; out = base + W2_OFF;
  } else if (r < 10240){               // w3: [64][64]
    int q = r - 6144; k = q >> 6; c = q & 63; NT = 4;
    val = rw3[l*4096 + q]; out = base + W3_OFF;
  } else if (r < 26624){               // w4: [64][256]
    int q = r - 10240; k = q >> 8; c = q & 255; NT = 16;
    val = rw4[l*16384 + q]; out = base + W4_OFF;
  } else {                             // mix: [256][64]
    int q = r - 26624; k = q >> 6; c = q & 63; NT = 4;
    val = mixw[l*16384 + q]; out = base + MIX_OFF;
  }
  int nt = c >> 4, n = c & 15, kstep = k >> 5, kg = (k >> 3) & 1, s = (k & 7) | (((k >> 4) & 1) << 3);
  out[(size_t)(((kstep*NT + nt) << 5) + (kg << 4) + n)*16 + s] = (_Float16)val;
}

// =====================  h0 = embed_w[species]  =====================
__global__ void tace_embed(const int* __restrict__ species, const float* __restrict__ ew,
                           float* __restrict__ h0, int N){
  int t = blockIdx.x*blockDim.x + threadIdx.x;
  if (t >= N*C_CH) return;
  int n = t >> 6, c = t & 63;
  h0[t] = ew[species[n]*C_CH + c];
}

// ============  edge kernel: geometry + radial-MLP (WMMA) + TP scatter  ============
__launch_bounds__(256)
__global__ void tace_edge(const int* __restrict__ senders, const int* __restrict__ recvs,
                          const float* __restrict__ ev, const float* __restrict__ h0,
                          const _Float16* __restrict__ wbank,  // this layer's swizzled bank
                          float* __restrict__ Aacc, int E)
{
  __shared__ _Float16 sRb[8][16*32];   // rbf tile (K padded to 32)      1 KB/wave
  __shared__ _Float16 sA0[8][16*64];   // activation ping                2 KB/wave
  __shared__ _Float16 sA1[8][16*64];   // activation pong                2 KB/wave
  __shared__ float    sY [8][16*DIMS]; // angular basis                  2.5 KB/wave
  __shared__ int      sRecv[8][16];
  __shared__ int      sSend[8][16];

  int tid = threadIdx.x;
  int wid = tid >> 5, lane = tid & 31;
  int e0 = (blockIdx.x*8 + wid)*16;
  if (e0 >= E) return;                 // wave-uniform

  _Float16* rb = sRb[wid];
  _Float16* a0 = sA0[wid];
  _Float16* a1 = sA1[wid];
  float*    yT = sY[wid];
  int*      rT = sRecv[wid];
  int*      sT = sSend[wid];

  const _Float16* w1s = wbank + W1_OFF;
  const _Float16* w2s = wbank + W2_OFF;
  const _Float16* w3s = wbank + W3_OFF;
  const _Float16* w4s = wbank + W4_OFF;

  // ---- per-edge geometry: rbf (K=8) + angular basis (40 dims), lanes 0-15 ----
  if (lane < 16){
    int e = e0 + lane; if (e >= E) e = E - 1;
    float vx = ev[3*e], vy = ev[3*e+1], vz = ev[3*e+2];
    float r  = sqrtf(vx*vx + vy*vy + vz*vz + 1e-12f);
    float x  = r * 0.2f;                                  // r / cutoff
    float x2 = x*x, x4 = x2*x2, x5 = x4*x, x6 = x5*x, x7 = x6*x;
    float env = 1.f - 21.f*x5 + 35.f*x6 - 15.f*x7;        // poly cutoff, p=5
    env = (x < 1.f) ? env : 0.f;
    float ir  = __builtin_amdgcn_rcpf(r);
    float pref = 0.632455532034f * ir * env;              // sqrt(2/cutoff)/r * env
    const float PI = 3.14159265358979f;
    v8h rbv;
#pragma unroll
    for (int k = 0; k < 8; ++k) rbv[k] = (_Float16)(pref * __sinf((float)(k+1)*PI*x));
    v8h z;
#pragma unroll
    for (int k = 0; k < 8; ++k) z[k] = (_Float16)0.f;
    v8h* rrow = (v8h*)(rb + lane*32);
    rrow[0] = rbv; rrow[1] = z; rrow[2] = z; rrow[3] = z;  // 4x ds_store_b128

    float rh[3] = {vx*ir, vy*ir, vz*ir};
    float yl[DIMS];
    yl[0] = 1.f;
    yl[1] = 1.73205080757f*rh[0]; yl[2] = 1.73205080757f*rh[1]; yl[3] = 1.73205080757f*rh[2];
#pragma unroll
    for (int i = 0; i < 3; ++i)
#pragma unroll
      for (int j = 0; j < 3; ++j)
        yl[4 + i*3 + j] = 2.2360679775f*rh[i]*rh[j];
#pragma unroll
    for (int i = 0; i < 3; ++i)
#pragma unroll
      for (int j = 0; j < 3; ++j)
#pragma unroll
        for (int k = 0; k < 3; ++k)
          yl[13 + i*9 + j*3 + k] = 2.64575131106f*rh[i]*rh[j]*rh[k];
    float* y = yT + lane*DIMS;
#pragma unroll
    for (int i = 0; i < 10; ++i)                           // 10x ds_store_b128
      *(float4*)(y + i*4) = make_float4(yl[4*i], yl[4*i+1], yl[4*i+2], yl[4*i+3]);
    rT[lane] = recvs[e];
    sT[lane] = senders[e];
  }

  // ---- GEMM1: [16,32(rbf pad)] @ w1[32,64] -> silu -> a0 ----
  {
    v16h a = fragA_lds(rb, 32, 0, lane);
#pragma unroll
    for (int nt = 0; nt < 4; ++nt){
      v8f acc = {};
      acc = WMMA_F16(a, fragB_sw(w1s, nt, lane), acc);
      int col = nt*16 + (lane & 15);
#pragma unroll
      for (int v = 0; v < 8; ++v){
        int row = v + ((lane >> 4) << 3);
        a0[row*64 + col] = (_Float16)silu_f(acc[v]);
      }
    }
  }
  // ---- GEMM2: a0 @ w2 -> silu -> a1 ----
  {
    v16h aL = fragA_lds(a0, 64, 0, lane);
    v16h aH = fragA_lds(a0, 64, 32, lane);
#pragma unroll
    for (int nt = 0; nt < 4; ++nt){
      v8f acc = {};
      acc = WMMA_F16(aL, fragB_sw(w2s, nt,     lane), acc);
      acc = WMMA_F16(aH, fragB_sw(w2s, 4 + nt, lane), acc);
      int col = nt*16 + (lane & 15);
#pragma unroll
      for (int v = 0; v < 8; ++v){
        int row = v + ((lane >> 4) << 3);
        a1[row*64 + col] = (_Float16)silu_f(acc[v]);
      }
    }
  }
  // ---- GEMM3: a1 @ w3 -> silu -> a0 ----
  {
    v16h aL = fragA_lds(a1, 64, 0, lane);
    v16h aH = fragA_lds(a1, 64, 32, lane);
#pragma unroll
    for (int nt = 0; nt < 4; ++nt){
      v8f acc = {};
      acc = WMMA_F16(aL, fragB_sw(w3s, nt,     lane), acc);
      acc = WMMA_F16(aH, fragB_sw(w3s, 4 + nt, lane), acc);
      int col = nt*16 + (lane & 15);
#pragma unroll
      for (int v = 0; v < 8; ++v){
        int row = v + ((lane >> 4) << 3);
        a0[row*64 + col] = (_Float16)silu_f(acc[v]);
      }
    }
  }
  // ---- GEMM4: a0 @ w4[64,256], scatter straight from WMMA accumulators ----
  {
    v16h aL = fragA_lds(a0, 64, 0, lane);
    v16h aH = fragA_lds(a0, 64, 32, lane);
    const float inv16 = 0.0625f;
#pragma unroll 1
    for (int nt = 0; nt < 16; ++nt){
      v8f acc = {};
      acc = WMMA_F16(aL, fragB_sw(w4s, nt,      lane), acc);
      acc = WMMA_F16(aH, fragB_sw(w4s, 16 + nt, lane), acc);
      int ri = nt >> 2;
      int cc = (nt & 3)*16 + (lane & 15);                 // channel 0..63 within rank
      int dlo = (ri == 0) ? 0 : (ri == 1) ? 1 : (ri == 2) ? 4 : 13;
      int dhi = (ri == 0) ? 1 : (ri == 1) ? 4 : (ri == 2) ? 13 : DIMS;
#pragma unroll
      for (int v = 0; v < 8; ++v){
        int er = v + ((lane >> 4) << 3);
        if (e0 + er >= E) continue;
        float hjv = h0[(size_t)sT[er]*C_CH + cc];         // sender gather (L2-resident)
        float val = acc[v] * hjv * inv16;
        float* Ab = Aacc + (size_t)rT[er]*(C_CH*DIMS) + cc;
        const float* ye = yT + er*DIMS;
        for (int dd = dlo; dd < dhi; ++dd)
          atomicAdd(Ab + dd*C_CH, val * ye[dd]);          // A layout [N][dim][ch]
      }
    }
  }
}

// ========  per-node rank invariants: U[N][4*64] f16 (s0 and correlation-2)  ========
__global__ void tace_node_u(const float* __restrict__ Aacc, _Float16* __restrict__ Uh, int N){
  int t = blockIdx.x*blockDim.x + threadIdx.x;
  if (t >= N*C_CH) return;
  int n = t >> 6, c = t & 63;
  const float* An = Aacc + (size_t)n*(C_CH*DIMS) + c;    // stride 64 over dim
  float s0 = An[0];
  float t1 = 0.f, t2 = 0.f, t3 = 0.f;
#pragma unroll
  for (int dd = 1;  dd < 4;  ++dd){ float a = An[dd*C_CH]; t1 += a*a; }
#pragma unroll
  for (int dd = 4;  dd < 13; ++dd){ float a = An[dd*C_CH]; t2 += a*a; }
#pragma unroll
  for (int dd = 13; dd < 40; ++dd){ float a = An[dd*C_CH]; t3 += a*a; }
  _Float16* u = Uh + (size_t)n*256 + c;
  u[0]   = (_Float16)s0;
  u[64]  = (_Float16)t1;
  u[128] = (_Float16)t2;
  u[192] = (_Float16)t3;
}

// ====  mix GEMM [N,256]@[256,64] (WMMA) + energy readout + h0 update  ====
__launch_bounds__(256)
__global__ void tace_node_mix(const _Float16* __restrict__ Uh, const _Float16* __restrict__ mixs,
                              const float* __restrict__ readw, float* __restrict__ h0next,
                              float* __restrict__ energy, int N, int addEnergy)
{
  int tid = threadIdx.x;
  int wid = tid >> 5, lane = tid & 31;
  int n0 = (blockIdx.x*8 + wid)*16;
  if (n0 >= N) return;                 // wave-uniform

  int row = n0 + (lane & 15); if (row >= N) row = N - 1;
  const _Float16* up = Uh + (size_t)row*256;

  float er[8];
#pragma unroll
  for (int v = 0; v < 8; ++v) er[v] = 0.f;

#pragma unroll
  for (int nt = 0; nt < 4; ++nt){
    v8f acc = {};
#pragma unroll
    for (int ks = 0; ks < 8; ++ks){
      v16h a = fragA_g16(up, ks*32, lane);
      v16h b = fragB_sw(mixs, ks*4 + nt, lane);
      acc = WMMA_F16(a, b, acc);
    }
    int col = nt*16 + (lane & 15);
    float rwv = readw[col];
#pragma unroll
    for (int v = 0; v < 8; ++v){
      int n = n0 + v + ((lane >> 4) << 3);
      if (n < N) h0next[(size_t)n*C_CH + col] = acc[v];
      er[v] += acc[v]*rwv;
    }
  }
  // dot over 64 channels: reduce across each 16-lane group (wave32 shuffles)
#pragma unroll
  for (int v = 0; v < 8; ++v){
    er[v] += __shfl_xor(er[v], 1, 32);
    er[v] += __shfl_xor(er[v], 2, 32);
    er[v] += __shfl_xor(er[v], 4, 32);
    er[v] += __shfl_xor(er[v], 8, 32);
  }
  if ((lane & 15) == 0){
#pragma unroll
    for (int v = 0; v < 8; ++v){
      int n = n0 + v + ((lane >> 4) << 3);
      if (n < N){
        if (addEnergy) energy[n] += er[v];
        else           energy[n]  = er[v];
      }
    }
  }
}

// ==============================  host launcher  ==============================
extern "C" void kernel_launch(void* const* d_in, const int* in_sizes, int n_in,
                              void* d_out, int out_size, void* d_ws, size_t ws_size,
                              hipStream_t stream) {
  const int*   species = (const int*)  d_in[0];
  const int*   senders = (const int*)  d_in[1];
  const int*   recvs   = (const int*)  d_in[2];
  const float* edgevec = (const float*)d_in[3];
  const float* embedw  = (const float*)d_in[4];
  const float* rw1     = (const float*)d_in[5];   // [2][8][64]
  const float* rw2     = (const float*)d_in[6];   // [2][64][64]
  const float* rw3     = (const float*)d_in[7];   // [2][64][64]
  const float* rw4     = (const float*)d_in[8];   // [2][64][256]
  const float* mixw    = (const float*)d_in[9];   // [2][4][64][64]
  const float* readw   = (const float*)d_in[10];  // [2][64]

  const int N = in_sizes[0];
  const int E = in_sizes[1];

  // workspace carve-up (all offsets 32B-aligned)
  char* ws = (char*)d_ws;
  size_t nc4   = (size_t)N*C_CH*sizeof(float);
  size_t asz   = (size_t)N*C_CH*DIMS*sizeof(float);
  size_t usz   = (size_t)N*256*sizeof(_Float16);
  float*     h0a  = (float*)ws;
  float*     h0b  = (float*)(ws + nc4);
  float*     Aacc = (float*)(ws + 2*nc4);
  _Float16*  Uh   = (_Float16*)(ws + 2*nc4 + asz);
  _Float16*  bank = (_Float16*)(ws + 2*nc4 + asz + usz);  // 2*PER_L halfs

  float* energy = (float*)d_out;

  int thr = 256;
  tace_prep<<<(2*PER_L + thr - 1)/thr, thr, 0, stream>>>(rw1, rw2, rw3, rw4, mixw, bank);

  int gEmb = (N*C_CH + thr - 1)/thr;
  tace_embed<<<gEmb, thr, 0, stream>>>(species, embedw, h0a, N);

  float* hcur = h0a;
  float* hnxt = h0b;
  int gEdge = (E + 127)/128;             // 8 waves x 16 edges per block
  int gMix  = (N + 127)/128;

  for (int l = 0; l < 2; ++l){
    hipMemsetAsync(Aacc, 0, asz, stream);
    tace_edge<<<gEdge, 256, 0, stream>>>(senders, recvs, edgevec, hcur,
                                         bank + (size_t)l*PER_L, Aacc, E);
    tace_node_u<<<gEmb, thr, 0, stream>>>(Aacc, Uh, N);
    tace_node_mix<<<gMix, 256, 0, stream>>>(Uh,
                                            bank + (size_t)l*PER_L + MIX_OFF,
                                            readw + (size_t)l*64,
                                            hnxt, energy, N, l > 0);
    float* t = hcur; hcur = hnxt; hnxt = t;
  }
}
